// DocRedModel_40656160424564
// MI455X (gfx1250) — matrix operand order repaired and checked by reading, not compiled, not run on hardware
//
#include <hip/hip_runtime.h>
#include <math.h>

// ---------------- problem constants ----------------
constexpr int NB  = 4;     // batches
constexpr int L   = 1024;  // sequence length
constexpr int HD  = 1024;  // hidden
constexpr int AH  = 16;    // attn heads
constexpr int MM  = 128;   // mentions
constexpr int NE  = 32;    // entities
constexpr int R   = 992;   // relation pairs per batch
constexpr int EMB = 768;
constexpr int CC  = 97;    // classes
constexpr int CP  = 112;   // classes padded to 7*16
constexpr int K2  = 2 * HD;      // 2048, concat K
constexpr int KBIL= EMB * 64;    // 49152, bilinear K
constexpr int NROW= NB * R;      // 3968 rows
constexpr int KCH = 4;           // K-split chunks for bilinear GEMM
constexpr int LDS_STRIDE = 40;   // halves per staged column (32 + 8 pad)

// ---------------- CDNA5 WMMA types ----------------
typedef __attribute__((ext_vector_type(16))) __bf16    v16bf;
typedef __attribute__((ext_vector_type(8)))  float     v8f;
typedef __attribute__((ext_vector_type(4)))  unsigned  u32x4;

struct bfx2 { u32x4 lo, hi; };

static __device__ __forceinline__ unsigned short f2bf(float f) {
  unsigned u = __builtin_bit_cast(unsigned, f);
  u += 0x7fffu + ((u >> 16) & 1u);
  return (unsigned short)(u >> 16);
}
static __device__ __forceinline__ float bf2f(unsigned short s) {
  unsigned u = ((unsigned)s) << 16;
  return __builtin_bit_cast(float, u);
}

// A-fragment: 16-bit A 16x32 layout.  lane<16 holds K {0..7, 16..23},
// lane>=16 holds K {8..15, 24..31}; caller passes p = rowbase + kstep + grp*8.
static __device__ __forceinline__ v16bf ld_a16(const unsigned short* p) {
  bfx2 t;
  t.lo = *(const u32x4*)(p);
  t.hi = *(const u32x4*)(p + 16);
  return __builtin_bit_cast(v16bf, t);
}
// B-fragment from an LDS-staged column (K-contiguous): lane<16 holds K {0..7, 16..23}, lane>=16 holds K {8..15, 24..31}
static __device__ __forceinline__ v16bf ld_b_lds(const unsigned short* p) {
  bfx2 t;
  t.lo = *(const u32x4*)(p);
  t.hi = *(const u32x4*)(p + 16);
  return __builtin_bit_cast(v16bf, t);
}

static __device__ __forceinline__ v8f wmma_bf16(v16bf a, v16bf b, v8f c) {
  return __builtin_amdgcn_wmma_f32_16x16x32_bf16(
      /*neg_a=*/false, a, /*neg_b=*/false, b,
      /*c_mod=*/(short)0, c, /*reuse_a=*/false, /*reuse_b=*/false);
}

// scale a packed pair of bf16 by h, repack (for on-the-fly bilinear A tile)
static __device__ __forceinline__ unsigned mul_pk(unsigned pk, float h) {
  unsigned lo = (pk & 0xffffu) << 16;
  unsigned hi = pk & 0xffff0000u;
  float a = __builtin_bit_cast(float, lo) * h;
  float b = __builtin_bit_cast(float, hi) * h;
  unsigned ua = __builtin_bit_cast(unsigned, a); ua += 0x7fffu + ((ua >> 16) & 1u);
  unsigned ub = __builtin_bit_cast(unsigned, b); ub += 0x7fffu + ((ub >> 16) & 1u);
  return (ua >> 16) | (ub & 0xffff0000u);
}

// ---------------- stage 0: transpose + f32->bf16 convert ----------------
__global__ void tconv_kernel(const float* __restrict__ src,
                             unsigned short* __restrict__ dst,
                             int K, int N, int Npad,
                             long long sBatch, long long dBatch) {
  long long z = blockIdx.z;
  int idx = blockIdx.x * 256 + threadIdx.x;
  long long total = (long long)K * Npad;
  if (idx >= total) return;
  int k = idx / Npad;
  int n = idx - k * Npad;
  float v = (n < N) ? src[sBatch * z + (long long)k * N + n] : 0.0f;
  dst[dBatch * z + (long long)n * K + k] = f2bf(v);
}

// ---------------- stage 1a: per-entity logsumexp pooling ----------------
__global__ void ent_pool_kernel(const float* __restrict__ ent,
                                const int* __restrict__ ids,
                                unsigned short* __restrict__ embbf) {
  __shared__ int sids[MM];
  const int b = blockIdx.z, e = blockIdx.y;
  const int h = blockIdx.x * 256 + threadIdx.x;
  if (threadIdx.x < MM) sids[threadIdx.x] = ids[b * MM + threadIdx.x];
  __syncthreads();
  float mx = -INFINITY, s = 0.0f; int cnt = 0;
  for (int m = 0; m < MM; ++m) {
    if (sids[m] == e) {
      float x = ent[((size_t)(b * MM + m)) * HD + h];
      ++cnt;
      if (x > mx) { s = s * __expf(mx - x) + 1.0f; mx = x; }
      else        { s += __expf(x - mx); }
    }
  }
  float v = (cnt > 0) ? (mx + __logf(s)) : 0.0f;
  embbf[((size_t)(b * NE + e)) * HD + h] = f2bf(v);
}

// ---------------- stage 1b: per-entity mean attention ----------------
__global__ void ent_attn_kernel(const float* __restrict__ attn,
                                const int* __restrict__ ids,
                                float* __restrict__ attnE) {
  __shared__ int   sids[MM];
  __shared__ float sinv[NE];
  __shared__ float acc[NE][256];
  const int b = blockIdx.z, a = blockIdx.y;
  const int t = threadIdx.x;
  const int l = blockIdx.x * 256 + t;
  if (t < MM) sids[t] = ids[b * MM + t];
  __syncthreads();
  if (t < NE) {
    int c = 0;
    for (int m = 0; m < MM; ++m) c += (sids[m] == t);
    sinv[t] = (c > 0) ? (1.0f / (float)c) : 0.0f;
  }
  #pragma unroll
  for (int e = 0; e < NE; ++e) acc[e][t] = 0.0f;
  __syncthreads();
  for (int m = 0; m < MM; ++m) {
    int e = sids[m];
    acc[e][t] += attn[(((size_t)(b * AH + a)) * MM + m) * L + l];
  }
  __syncthreads();
  #pragma unroll
  for (int e = 0; e < NE; ++e)
    attnE[(((size_t)(b * NE + e)) * AH + a) * L + l] = acc[e][t] * sinv[e];
}

// ---------------- stage 1c: pair weights w (normalized) ----------------
__global__ void w_kernel(const float* __restrict__ attnE,
                         const int* __restrict__ hts,
                         unsigned short* __restrict__ wbf) {
  __shared__ float red[256];
  const int b = blockIdx.y, r = blockIdx.x, t = threadIdx.x;
  const int he = hts[(b * R + r) * 2 + 0];
  const int te = hts[(b * R + r) * 2 + 1];
  const float* ph = attnE + ((size_t)(b * NE + he)) * AH * L;
  const float* pt = attnE + ((size_t)(b * NE + te)) * AH * L;
  float wl[L / 256];
  float ls = 0.0f;
  #pragma unroll
  for (int li = 0; li < L / 256; ++li) {
    int l = t + li * 256;
    float s = 0.0f;
    #pragma unroll
    for (int a = 0; a < AH; ++a) s += ph[a * L + l] * pt[a * L + l];
    wl[li] = s * (1.0f / (float)AH);
    ls += wl[li];
  }
  red[t] = ls; __syncthreads();
  for (int s = 128; s > 0; s >>= 1) {
    if (t < s) red[t] += red[t + s];
    __syncthreads();
  }
  float inv = 1.0f / (red[0] + 1e-5f);
  #pragma unroll
  for (int li = 0; li < L / 256; ++li)
    wbf[((size_t)(b * R + r)) * L + (t + li * 256)] = f2bf(wl[li] * inv);
}

// ---------------- GEMM 1: rel = w @ seq  (per batch 992x1024 @ 1024x1024) ----
// 256-thread block = 8 waves: each wave one 16-row M-tile, block shares a
// 128-col B panel double-buffered in LDS (8x B-traffic reduction).
__global__ void __launch_bounds__(256)
gemm_rel_kernel(const unsigned short* __restrict__ wbf,
                const unsigned short* __restrict__ seqT,
                unsigned short* __restrict__ relbf) {
  __shared__ unsigned short Bs[2][128 * LDS_STRIDE];
  const int b  = blockIdx.z;
  const int mb = blockIdx.x & 7;   // 8 M-blocks (last partially valid)
  const int nb = blockIdx.x >> 3;  // 8 N-blocks
  const int w  = threadIdx.x >> 5;
  const int lane = threadIdx.x & 31, l15 = lane & 15, grp = lane >> 4;
  const int MT = R / 16;           // 62
  const int mtile = mb * 8 + w;
  const bool valid = (mtile < MT);
  const int mtc = valid ? mtile : 0;
  const unsigned short* arow  = wbf  + (size_t)(b * R + mtc * 16 + l15) * L;
  const unsigned short* bbase = seqT + (size_t)b * HD * L + (size_t)(nb * 128) * L;
  const int c0 = threadIdx.x;
  const int col0 = c0 >> 2, q0 = c0 & 3;          // chunk 0
  const int col1 = (c0 + 256) >> 2, q1 = c0 & 3;  // chunk 1

  *(u32x4*)(&Bs[0][col0 * LDS_STRIDE + q0 * 8]) =
      *(const u32x4*)(bbase + (size_t)col0 * L + q0 * 8);
  *(u32x4*)(&Bs[0][col1 * LDS_STRIDE + q1 * 8]) =
      *(const u32x4*)(bbase + (size_t)col1 * L + q1 * 8);
  __syncthreads();

  v8f acc[8];
  #pragma unroll
  for (int i = 0; i < 8; ++i) acc[i] = (v8f)0.0f;

  const int NSt = L / 32;
  for (int s = 0; s < NSt; ++s) {
    const int kbase = s * 32;
    u32x4 pf0, pf1;
    if (s + 1 < NSt) {
      pf0 = *(const u32x4*)(bbase + (size_t)col0 * L + kbase + 32 + q0 * 8);
      pf1 = *(const u32x4*)(bbase + (size_t)col1 * L + kbase + 32 + q1 * 8);
    }
    v16bf av = ld_a16(arow + kbase + grp * 8);
    const unsigned short* bsb = Bs[s & 1];
    #pragma unroll
    for (int nt = 0; nt < 8; ++nt) {
      v16bf bv = ld_b_lds(bsb + (nt * 16 + l15) * LDS_STRIDE + grp * 8);
      acc[nt] = wmma_bf16(av, bv, acc[nt]);
    }
    if (s + 1 < NSt) {
      unsigned short* bd = Bs[(s + 1) & 1];
      *(u32x4*)(&bd[col0 * LDS_STRIDE + q0 * 8]) = pf0;
      *(u32x4*)(&bd[col1 * LDS_STRIDE + q1 * 8]) = pf1;
    }
    __syncthreads();
  }
  if (valid) {
    #pragma unroll
    for (int nt = 0; nt < 8; ++nt) {
      int n = nb * 128 + nt * 16 + l15;
      #pragma unroll
      for (int v = 0; v < 8; ++v) {
        int row = b * R + mtile * 16 + v + 8 * grp;
        relbf[(size_t)row * HD + n] = f2bf(acc[nt][v]);
      }
    }
  }
}

// ---------------- GEMM 2: hs/ts = tanh(concat(emb,rel) @ W + b) ------------
// Same LDS-shared-B scheme; A rows gathered from embbf (k<1024)/relbf (k>=1024).
__global__ void __launch_bounds__(256)
gemm_ht_kernel(const unsigned short* __restrict__ embbf,
               const unsigned short* __restrict__ relbf,
               const int* __restrict__ hts,
               const unsigned short* __restrict__ WT,   // [EMB][K2] bf16
               const float* __restrict__ bias,
               unsigned short* __restrict__ outbf,       // [NROW][EMB] bf16
               int sel) {
  __shared__ unsigned short Bs[2][128 * LDS_STRIDE];
  const int mb = blockIdx.x % 31;  // 31 M-blocks (248 tiles, all valid)
  const int nb = blockIdx.x / 31;  // 6 N-blocks
  const int w  = threadIdx.x >> 5;
  const int lane = threadIdx.x & 31, l15 = lane & 15, grp = lane >> 4;
  const int mtile = mb * 8 + w;
  const int row = mtile * 16 + l15;
  const int bb = row / R, rr = row - bb * R;
  const int eidx = hts[(bb * R + rr) * 2 + sel];
  const unsigned short* embrow = embbf + ((size_t)(bb * NE + eidx)) * HD;
  const unsigned short* relrow = relbf + ((size_t)row) * HD;
  const unsigned short* bbase  = WT + (size_t)(nb * 128) * K2;
  const int c0 = threadIdx.x;
  const int col0 = c0 >> 2, q0 = c0 & 3;
  const int col1 = (c0 + 256) >> 2, q1 = c0 & 3;

  *(u32x4*)(&Bs[0][col0 * LDS_STRIDE + q0 * 8]) =
      *(const u32x4*)(bbase + (size_t)col0 * K2 + q0 * 8);
  *(u32x4*)(&Bs[0][col1 * LDS_STRIDE + q1 * 8]) =
      *(const u32x4*)(bbase + (size_t)col1 * K2 + q1 * 8);
  __syncthreads();

  v8f acc[8];
  #pragma unroll
  for (int i = 0; i < 8; ++i) acc[i] = (v8f)0.0f;

  const int NSt = K2 / 32;
  for (int s = 0; s < NSt; ++s) {
    const int kbase = s * 32;
    u32x4 pf0, pf1;
    if (s + 1 < NSt) {
      pf0 = *(const u32x4*)(bbase + (size_t)col0 * K2 + kbase + 32 + q0 * 8);
      pf1 = *(const u32x4*)(bbase + (size_t)col1 * K2 + kbase + 32 + q1 * 8);
    }
    const unsigned short* ap = (kbase < HD) ? (embrow + kbase) : (relrow + (kbase - HD));
    v16bf av = ld_a16(ap + grp * 8);
    const unsigned short* bsb = Bs[s & 1];
    #pragma unroll
    for (int nt = 0; nt < 8; ++nt) {
      v16bf bv = ld_b_lds(bsb + (nt * 16 + l15) * LDS_STRIDE + grp * 8);
      acc[nt] = wmma_bf16(av, bv, acc[nt]);
    }
    if (s + 1 < NSt) {
      unsigned short* bd = Bs[(s + 1) & 1];
      *(u32x4*)(&bd[col0 * LDS_STRIDE + q0 * 8]) = pf0;
      *(u32x4*)(&bd[col1 * LDS_STRIDE + q1 * 8]) = pf1;
    }
    __syncthreads();
  }
  #pragma unroll
  for (int nt = 0; nt < 8; ++nt) {
    int n = nb * 128 + nt * 16 + l15;
    float bn = bias[n];
    #pragma unroll
    for (int v = 0; v < 8; ++v) {
      int orow = mtile * 16 + v + 8 * grp;
      outbf[(size_t)orow * EMB + n] = f2bf(tanhf(acc[nt][v] + bn));
    }
  }
}

// ---------------- GEMM 3: bilinear head (bl built on the fly) --------------
// out_part[ch][n,c] = sum_{k in chunk} hs[n,gi]*ts[n,g*64+j]*Wt[c][k]
// 4-way K-split for parallelism; 8 waves share the 112-col B panel via LDS.
__global__ void __launch_bounds__(256)
gemm_bil_kernel(const unsigned short* __restrict__ hsbf,
                const unsigned short* __restrict__ tsbf,
                const unsigned short* __restrict__ bilWT, // [CP][KBIL] bf16
                float* __restrict__ part) {               // [KCH][NROW][CP]
  __shared__ unsigned short Bs[2][CP * LDS_STRIDE];
  const int mb = blockIdx.x % 31;  // 31 M-blocks
  const int ch = blockIdx.x / 31;  // K-chunk 0..3
  const int w  = threadIdx.x >> 5;
  const int lane = threadIdx.x & 31, l15 = lane & 15, grp = lane >> 4;
  const int mtile = mb * 8 + w;
  const int row = mtile * 16 + l15;
  const unsigned short* hsrow = hsbf + (size_t)row * EMB;
  const unsigned short* tsrow = tsbf + (size_t)row * EMB;
  const int k0 = ch * (KBIL / KCH);
  const int c0 = threadIdx.x;
  const int col0 = c0 >> 2, q0 = c0 & 3;
  const int col1 = (c0 + 256) >> 2, q1 = c0 & 3;
  const bool f1 = (c0 + 256) < CP * 4;   // 448 chunks total

  *(u32x4*)(&Bs[0][col0 * LDS_STRIDE + q0 * 8]) =
      *(const u32x4*)(bilWT + (size_t)col0 * KBIL + k0 + q0 * 8);
  if (f1)
    *(u32x4*)(&Bs[0][col1 * LDS_STRIDE + q1 * 8]) =
        *(const u32x4*)(bilWT + (size_t)col1 * KBIL + k0 + q1 * 8);
  __syncthreads();

  v8f acc[7];
  #pragma unroll
  for (int i = 0; i < 7; ++i) acc[i] = (v8f)0.0f;

  const int NSt = (KBIL / KCH) / 32;     // 384 K-steps of 32
  for (int s = 0; s < NSt; ++s) {
    const int kb = (k0 >> 5) + s;        // global K-step index
    u32x4 pf0, pf1;
    if (s + 1 < NSt) {
      pf0 = *(const u32x4*)(bilWT + (size_t)col0 * KBIL + k0 + (s + 1) * 32 + q0 * 8);
      if (f1)
        pf1 = *(const u32x4*)(bilWT + (size_t)col1 * KBIL + k0 + (s + 1) * 32 + q1 * 8);
    }
    // build A tile on the fly: A[m, kk] = hs[m, gi] * ts[m, g*64 + j0 + kk]
    const int gi = kb >> 1;
    const int g  = gi >> 6;
    const int j0 = (kb & 1) << 5;
    float hval = bf2f(hsrow[gi]);
    const unsigned short* tp = tsrow + (g << 6) + j0 + grp * 8;
    u32x4 lo = *(const u32x4*)tp;
    u32x4 hi = *(const u32x4*)(tp + 16);
    bfx2 t;
    #pragma unroll
    for (int q = 0; q < 4; ++q) { t.lo[q] = mul_pk(lo[q], hval); t.hi[q] = mul_pk(hi[q], hval); }
    v16bf av = __builtin_bit_cast(v16bf, t);
    const unsigned short* bsb = Bs[s & 1];
    #pragma unroll
    for (int nt = 0; nt < 7; ++nt) {
      v16bf bv = ld_b_lds(bsb + (nt * 16 + l15) * LDS_STRIDE + grp * 8);
      acc[nt] = wmma_bf16(av, bv, acc[nt]);
    }
    if (s + 1 < NSt) {
      unsigned short* bd = Bs[(s + 1) & 1];
      *(u32x4*)(&bd[col0 * LDS_STRIDE + q0 * 8]) = pf0;
      if (f1) *(u32x4*)(&bd[col1 * LDS_STRIDE + q1 * 8]) = pf1;
    }
    __syncthreads();
  }
  #pragma unroll
  for (int nt = 0; nt < 7; ++nt) {
    int n = nt * 16 + l15;
    #pragma unroll
    for (int v = 0; v < 8; ++v) {
      int orow = mtile * 16 + v + 8 * grp;
      part[((size_t)ch * NROW + orow) * CP + n] = acc[nt][v];
    }
  }
}

// ---------------- bilinear partial reduction + bias ----------------
__global__ void bil_reduce_kernel(const float* __restrict__ part,
                                  const float* __restrict__ bilb,
                                  float* __restrict__ out) {
  int idx = blockIdx.x * 256 + threadIdx.x;
  if (idx >= NROW * CC) return;
  int row = idx / CC, c = idx - row * CC;
  float s = bilb[c];
  #pragma unroll
  for (int ch = 0; ch < KCH; ++ch)
    s += part[((size_t)ch * NROW + row) * CP + c];
  out[idx] = s;
}

// ---------------- host-side launcher ----------------
extern "C" void kernel_launch(void* const* d_in, const int* in_sizes, int n_in,
                              void* d_out, int out_size, void* d_ws, size_t ws_size,
                              hipStream_t stream) {
  (void)in_sizes; (void)n_in; (void)out_size; (void)ws_size;
  const float* seq   = (const float*)d_in[0];
  const float* ent   = (const float*)d_in[1];
  const float* attn  = (const float*)d_in[2];
  const int*   ids   = (const int*)  d_in[3];
  const int*   hts   = (const int*)  d_in[4];
  const float* headW = (const float*)d_in[5];
  const float* headb = (const float*)d_in[6];
  const float* tailW = (const float*)d_in[7];
  const float* tailb = (const float*)d_in[8];
  const float* bilW  = (const float*)d_in[9];
  const float* bilb  = (const float*)d_in[10];
  float* out = (float*)d_out;

  // workspace carve-out (all sizes multiples of 256B)
  char* ws = (char*)d_ws;
  size_t off = 0;
  auto take = [&](size_t bytes) -> char* { char* p = ws + off; off += bytes; return p; };
  unsigned short* seqT   = (unsigned short*)take((size_t)NB * HD * L * 2);
  unsigned short* WheadT = (unsigned short*)take((size_t)EMB * K2 * 2);
  unsigned short* WtailT = (unsigned short*)take((size_t)EMB * K2 * 2);
  unsigned short* bilWT  = (unsigned short*)take((size_t)CP * KBIL * 2);
  unsigned short* embbf  = (unsigned short*)take((size_t)NB * NE * HD * 2);
  float*          attnE  = (float*)         take((size_t)NB * NE * AH * L * 4);
  unsigned short* wbf    = (unsigned short*)take((size_t)NB * R * L * 2);
  unsigned short* relbf  = (unsigned short*)take((size_t)NROW * HD * 2);
  unsigned short* hsbf   = (unsigned short*)take((size_t)NROW * EMB * 2);
  unsigned short* tsbf   = (unsigned short*)take((size_t)NROW * EMB * 2);
  float*          part   = (float*)         take((size_t)KCH * NROW * CP * 4);

  // stage 0: weight / seq transposition into bf16 K-contiguous layouts
  tconv_kernel<<<dim3((L * HD + 255) / 256, 1, NB), 256, 0, stream>>>(
      seq, seqT, L, HD, HD, (long long)L * HD, (long long)HD * L);
  tconv_kernel<<<dim3((K2 * EMB + 255) / 256, 1, 1), 256, 0, stream>>>(
      headW, WheadT, K2, EMB, EMB, 0, 0);
  tconv_kernel<<<dim3((K2 * EMB + 255) / 256, 1, 1), 256, 0, stream>>>(
      tailW, WtailT, K2, EMB, EMB, 0, 0);
  tconv_kernel<<<dim3((KBIL * CP + 255) / 256, 1, 1), 256, 0, stream>>>(
      bilW, bilWT, KBIL, CC, CP, 0, 0);

  // stage 1: pooling
  ent_pool_kernel<<<dim3(HD / 256, NE, NB), 256, 0, stream>>>(ent, ids, embbf);
  ent_attn_kernel<<<dim3(L / 256, AH, NB), 256, 0, stream>>>(attn, ids, attnE);
  w_kernel<<<dim3(R, NB), 256, 0, stream>>>(attnE, hts, wbf);

  // stage 2: rel = w @ seq  (WMMA, LDS-shared B)
  gemm_rel_kernel<<<dim3(64, 1, NB), 256, 0, stream>>>(wbf, seqT, relbf);

  // stage 3: head / tail projections (WMMA, fused gather + tanh)
  gemm_ht_kernel<<<31 * (EMB / 128), 256, 0, stream>>>(
      embbf, relbf, hts, WheadT, headb, hsbf, 0);
  gemm_ht_kernel<<<31 * (EMB / 128), 256, 0, stream>>>(
      embbf, relbf, hts, WtailT, tailb, tsbf, 1);

  // stage 4: bilinear classifier (WMMA, on-the-fly A, K-split) + reduce
  gemm_bil_kernel<<<31 * KCH, 256, 0, stream>>>(hsbf, tsbf, bilWT, part);
  bil_reduce_kernel<<<(NROW * CC + 255) / 256, 256, 0, stream>>>(part, bilb, out);
}
